// ConformerLayer_37769942401602
// MI455X (gfx1250) — matrix operand order repaired and checked by reading, not compiled, not run on hardware
//
#include <hip/hip_runtime.h>
#include <hip/hip_bf16.h>
#include <cstdint>

// ---------------------------------------------------------------------------
// Conformer layer for MI455X (gfx1250): all matmuls via v_wmma_f32_16x16x32_f16
// with fully vectorized (b128) fragment loads from a canonical B^T layout.
// ---------------------------------------------------------------------------

#define D_MODEL 512
#define D_FF    2048
#define N_HEAD  8
#define D_HEAD  64
#define SEQ_T   1024
#define BATCH_B 4
#define N_TOK   (BATCH_B * SEQ_T)     // 4096
#define P_POS   2047
#define P_PAD   2048
#define KCONV   31
#define LN_EPS  1e-5f

typedef __attribute__((ext_vector_type(16))) _Float16 v16h;
typedef __attribute__((ext_vector_type(8)))  _Float16 v8h;
typedef __attribute__((ext_vector_type(8)))  float    v8f;

__device__ __forceinline__ v8f wmma_f16(v16h a, v16h b, v8f c) {
  return __builtin_amdgcn_wmma_f32_16x16x32_f16(false, a, false, b, (short)0, c,
                                                false, false);
}

__device__ __forceinline__ v16h cat8(v8h lo, v8h hi) {
  return __builtin_shufflevector(lo, hi, 0, 1, 2, 3, 4, 5, 6, 7,
                                 8, 9, 10, 11, 12, 13, 14, 15);
}

__device__ __forceinline__ float sigmf(float x) { return 1.f / (1.f + __expf(-x)); }

// ---------------------------------------------------------------------------
// f32 -> f16 convert with optional zero pad tail
// ---------------------------------------------------------------------------
__global__ void cvt_f16_kernel(const float* __restrict__ src,
                               _Float16* __restrict__ dst, int n, int npad) {
  int i = blockIdx.x * 256 + threadIdx.x;
  if (i < n)            dst[i] = (_Float16)src[i];
  else if (i < npad)    dst[i] = (_Float16)0.f;
}

// f32 KxN row-major -> f16 NxK row-major (B^T canonical layout)
__global__ void cvt_f16_t_kernel(const float* __restrict__ src,
                                 _Float16* __restrict__ dst, int K, int N) {
  int i = blockIdx.x * 256 + threadIdx.x;
  if (i >= K * N) return;
  int k = i / N, n = i - k * N;
  dst[(size_t)n * K + k] = (_Float16)src[i];
}

// ---------------------------------------------------------------------------
// Row LayerNorm over D=512 (one block per token row, 256 threads x 2 cols)
// ---------------------------------------------------------------------------
__global__ void ln_kernel(const float* __restrict__ x,
                          const float* __restrict__ g,
                          const float* __restrict__ b,
                          _Float16* __restrict__ out16,
                          float* __restrict__ out32) {
  int row = blockIdx.x, tid = threadIdx.x;
  const float* xr = x + (size_t)row * D_MODEL;
  float v0 = xr[tid], v1 = xr[tid + 256];
  __shared__ float red[256];
  red[tid] = v0 + v1; __syncthreads();
  for (int o = 128; o > 0; o >>= 1) {
    if (tid < o) red[tid] += red[tid + o];
    __syncthreads();
  }
  float mean = red[0] * (1.f / D_MODEL); __syncthreads();
  float d0 = v0 - mean, d1 = v1 - mean;
  red[tid] = d0 * d0 + d1 * d1; __syncthreads();
  for (int o = 128; o > 0; o >>= 1) {
    if (tid < o) red[tid] += red[tid + o];
    __syncthreads();
  }
  float inv = rsqrtf(red[0] * (1.f / D_MODEL) + LN_EPS);
  float y0 = d0 * inv * g[tid] + b[tid];
  float y1 = d1 * inv * g[tid + 256] + b[tid + 256];
  if (out16) { out16[(size_t)row * D_MODEL + tid] = (_Float16)y0;
               out16[(size_t)row * D_MODEL + tid + 256] = (_Float16)y1; }
  if (out32) { out32[(size_t)row * D_MODEL + tid] = y0;
               out32[(size_t)row * D_MODEL + tid + 256] = y1; }
}

// ---------------------------------------------------------------------------
// Batched WMMA GEMM: C = alpha * act(A @ B + bias) + residual
// A:  f16 MxK row-major (lda).
// BT: f16 NxK row-major (ldb) == B^T  -> every fragment load is 2x b128.
// Each wave computes a 16x64 C strip (4 accumulators, A fragment reused 4x).
// Fragment layouts per CDNA5 ISA 7.12.2 (wave32):
//   A 16x32:  lane m=lan, k = e + 8*gsel + (e>=8 ? 8 : 0)  -> two runs of 8
//   B 32x16:  lane n=lan, k = e + 8*gsel + (e>=8 ? 8 : 0)  -> two runs of 8
//   C 16x16:  lane n=lan, M = r + 8*gsel
// ---------------------------------------------------------------------------
__global__ void wmma_gemm_kernel(const _Float16* __restrict__ A, long long sA,
                                 const _Float16* __restrict__ BT, long long sB,
                                 const float* __restrict__ bias,
                                 const float* __restrict__ res, long long sR,
                                 float* __restrict__ C32, long long sC,
                                 _Float16* __restrict__ C16, long long sC16,
                                 int M, int N, int K,
                                 int lda, int ldb, int ldc,
                                 int act, float alpha) {
  int batch = blockIdx.y;
  const _Float16* Ab = A + (size_t)batch * sA;
  const _Float16* Bb = BT + (size_t)batch * sB;
  int wave = threadIdx.x >> 5, lane = threadIdx.x & 31;
  int stripsN = N >> 6;
  int strips  = (M >> 4) * stripsN;
  int t = blockIdx.x * 4 + wave;
  if (t >= strips) return;
  int tm = t / stripsN, tn = t % stripsN;
  int gsel = lane >> 4, lan = lane & 15;

  v8f acc[4];
#pragma unroll
  for (int j = 0; j < 4; ++j) acc[j] = (v8f){0.f,0.f,0.f,0.f,0.f,0.f,0.f,0.f};

  const _Float16* Arow = Ab + (size_t)(tm * 16 + lan) * lda + 8 * gsel;
  const _Float16* Bp[4];
#pragma unroll
  for (int j = 0; j < 4; ++j)
    Bp[j] = Bb + (size_t)(tn * 64 + j * 16 + lan) * ldb + 8 * gsel;

  for (int k0 = 0; k0 < K; k0 += 32) {
    __builtin_prefetch((const void*)(Arow + k0 + 128), 0, 3);
    __builtin_prefetch((const void*)(Bp[0] + k0 + 128), 0, 3);
    v16h a = cat8(*(const v8h*)(Arow + k0),
                  *(const v8h*)(Arow + k0 + 16));
#pragma unroll
    for (int j = 0; j < 4; ++j) {
      v16h b = cat8(*(const v8h*)(Bp[j] + k0),
                    *(const v8h*)(Bp[j] + k0 + 16));
      acc[j] = wmma_f16(a, b, acc[j]);
    }
  }

#pragma unroll
  for (int j = 0; j < 4; ++j) {
    int n = tn * 64 + j * 16 + lan;
    float bval = bias ? bias[n] : 0.f;
#pragma unroll
    for (int r = 0; r < 8; ++r) {
      int m = tm * 16 + r + 8 * gsel;
      float v = acc[j][r] + bval;
      if (act == 1) v = v * sigmf(v);               // SiLU
      v *= alpha;
      size_t ci = (size_t)m * ldc + n;
      if (res) v += res[(size_t)batch * sR + ci];
      if (C32) C32[(size_t)batch * sC + ci] = v;
      if (C16) C16[(size_t)batch * sC16 + ci] = (_Float16)v;
    }
  }
}

// ---------------------------------------------------------------------------
// bd term with fused rel-shift:  scores[q, j] += (q+pos_v)_q . p_{T-1+j-q}
// 16x64 strips of the (T x P_PAD, K=64) GEMM per (b,h); scatter-add shifted.
// ---------------------------------------------------------------------------
__global__ void bd_shift_kernel(const _Float16* __restrict__ qv,
                                const _Float16* __restrict__ p16,
                                float* __restrict__ scores) {
  const int T = SEQ_T;
  int bh = blockIdx.y, h = bh & (N_HEAD - 1);
  int wave = threadIdx.x >> 5, lane = threadIdx.x & 31;
  int stripsP = P_PAD >> 6;                 // 32
  int strips  = (T >> 4) * stripsP;         // 2048
  int t = blockIdx.x * 4 + wave;
  if (t >= strips) return;
  int tm = t / stripsP, tn = t % stripsP;
  int gsel = lane >> 4, lan = lane & 15;

  const _Float16* Arow = qv + (size_t)bh * T * D_HEAD +
                         (size_t)(tm * 16 + lan) * D_HEAD + 8 * gsel;
  const _Float16* Bp[4];
#pragma unroll
  for (int j = 0; j < 4; ++j)
    Bp[j] = p16 + (size_t)(tn * 64 + j * 16 + lan) * D_MODEL + h * D_HEAD +
            8 * gsel;

  v8f acc[4];
#pragma unroll
  for (int j = 0; j < 4; ++j) acc[j] = (v8f){0.f,0.f,0.f,0.f,0.f,0.f,0.f,0.f};

#pragma unroll
  for (int k0 = 0; k0 < D_HEAD; k0 += 32) {
    v16h a = cat8(*(const v8h*)(Arow + k0),
                  *(const v8h*)(Arow + k0 + 16));
#pragma unroll
    for (int j = 0; j < 4; ++j) {
      v16h b = cat8(*(const v8h*)(Bp[j] + k0),
                    *(const v8h*)(Bp[j] + k0 + 16));
      acc[j] = wmma_f16(a, b, acc[j]);
    }
  }

  float* srow = scores + (size_t)bh * T * T;
#pragma unroll
  for (int j = 0; j < 4; ++j) {
    int pidx = tn * 64 + j * 16 + lan;
#pragma unroll
    for (int r = 0; r < 8; ++r) {
      int q  = tm * 16 + r + 8 * gsel;
      int jj = pidx - (T - 1) + q;                 // rel_shift inverse
      if (jj >= 0 && jj < T) srow[(size_t)q * T + jj] += acc[j][r];
    }
  }
}

// ---------------------------------------------------------------------------
// Row softmax over T=1024 (scale fused), emit f16 attention weights
// ---------------------------------------------------------------------------
__global__ void softmax_kernel(const float* __restrict__ scores,
                               _Float16* __restrict__ attn16, float scale) {
  size_t row = blockIdx.x;
  const float* s = scores + row * SEQ_T;
  _Float16* a = attn16 + row * SEQ_T;
  int tid = threadIdx.x;
  float vals[4], m = -1e30f;
#pragma unroll
  for (int i = 0; i < 4; ++i) {
    vals[i] = s[tid + i * 256] * scale;
    m = fmaxf(m, vals[i]);
  }
  __shared__ float red[256];
  red[tid] = m; __syncthreads();
  for (int o = 128; o > 0; o >>= 1) {
    if (tid < o) red[tid] = fmaxf(red[tid], red[tid + o]);
    __syncthreads();
  }
  m = red[0]; __syncthreads();
  float sum = 0.f;
#pragma unroll
  for (int i = 0; i < 4; ++i) { vals[i] = __expf(vals[i] - m); sum += vals[i]; }
  red[tid] = sum; __syncthreads();
  for (int o = 128; o > 0; o >>= 1) {
    if (tid < o) red[tid] += red[tid + o];
    __syncthreads();
  }
  float inv = 1.f / red[0];
#pragma unroll
  for (int i = 0; i < 4; ++i) a[tid + i * 256] = (_Float16)(vals[i] * inv);
}

// ---------------------------------------------------------------------------
// (B,T,H*DK) f32 q/k/v  ->  f16 attention layouts:
//   qu/qv/k : (B,H,T,DK)      (A operand of ac GEMM / B^T of ac GEMM)
//   vT      : (B,H,DK,T)      (B^T of the attn@V GEMM)
// ---------------------------------------------------------------------------
__global__ void qkv_prep_kernel(const float* __restrict__ q32,
                                const float* __restrict__ k32,
                                const float* __restrict__ v32,
                                const float* __restrict__ pos_u,
                                const float* __restrict__ pos_v,
                                _Float16* __restrict__ qu, _Float16* __restrict__ qvv,
                                _Float16* __restrict__ k16, _Float16* __restrict__ vT16) {
  int i = blockIdx.x * 256 + threadIdx.x;
  if (i >= N_TOK * D_MODEL) return;
  int n = i >> 9, c = i & 511;
  int h = c >> 6, d = c & 63;
  int b = n >> 10, t = n & 1023;
  int bh = b * N_HEAD + h;
  size_t o = ((size_t)bh * SEQ_T + t) * D_HEAD + d;
  float qq = q32[i];
  qu[o]  = (_Float16)(qq + pos_u[c]);
  qvv[o] = (_Float16)(qq + pos_v[c]);
  k16[o] = (_Float16)k32[i];
  vT16[((size_t)bh * D_HEAD + d) * SEQ_T + t] = (_Float16)v32[i];
}

// (B,H,T,DK) f32 attention output -> (B,T,H*DK) f16
__global__ void ao_pack_kernel(const float* __restrict__ ao32,
                               _Float16* __restrict__ ao16) {
  int i = blockIdx.x * 256 + threadIdx.x;
  if (i >= N_TOK * D_MODEL) return;
  int n = i >> 9, c = i & 511;
  int h = c >> 6, d = c & 63;
  int b = n >> 10, t = n & 1023;
  size_t o = (((size_t)(b * N_HEAD + h) * SEQ_T) + t) * D_HEAD + d;
  ao16[i] = (_Float16)ao32[o];
}

// GLU: z (N x 1024) -> yg (N x 512)
__global__ void glu_kernel(const float* __restrict__ z, float* __restrict__ yg) {
  int i = blockIdx.x * 256 + threadIdx.x;
  if (i >= N_TOK * D_MODEL) return;
  int n = i >> 9, d = i & 511;
  float a = z[(size_t)n * 1024 + d];
  float g = z[(size_t)n * 1024 + 512 + d];
  yg[i] = a * sigmf(g);
}

// depthwise conv along time (31 taps, same-pad), per (b,t,d)
__global__ void dwconv_kernel(const float* __restrict__ yg,
                              const float* __restrict__ w,
                              const float* __restrict__ bias,
                              float* __restrict__ yc) {
  int i = blockIdx.x * 256 + threadIdx.x;
  if (i >= N_TOK * D_MODEL) return;
  int d = i & 511, n = i >> 9;
  int t = n & 1023, b = n >> 10;
  float acc = bias[d];
#pragma unroll
  for (int k = 0; k < KCONV; ++k) {
    int tt = t + k - (KCONV - 1) / 2;
    if (tt >= 0 && tt < SEQ_T)
      acc += yg[(((size_t)b * SEQ_T + tt) << 9) + d] * w[d * KCONV + k];
  }
  yc[i] = acc;
}

// BatchNorm stats over (b,t) per channel -> fused scale/shift
__global__ void bn_stats_kernel(const float* __restrict__ yc,
                                const float* __restrict__ g,
                                const float* __restrict__ b,
                                float* __restrict__ scale,
                                float* __restrict__ shift) {
  int d = blockIdx.x, tid = threadIdx.x;
  float s = 0.f, ss = 0.f;
  for (int n = tid; n < N_TOK; n += 256) {
    float v = yc[((size_t)n << 9) + d];
    s += v; ss += v * v;
  }
  __shared__ float r0[256], r1[256];
  r0[tid] = s; r1[tid] = ss; __syncthreads();
  for (int o = 128; o > 0; o >>= 1) {
    if (tid < o) { r0[tid] += r0[tid + o]; r1[tid] += r1[tid + o]; }
    __syncthreads();
  }
  if (tid == 0) {
    float mean = r0[0] * (1.f / N_TOK);
    float var  = r1[0] * (1.f / N_TOK) - mean * mean;
    float inv  = rsqrtf(var + LN_EPS) * g[d];
    scale[d] = inv;
    shift[d] = b[d] - mean * inv;
  }
}

// y = silu(yc*scale + shift) -> f16
__global__ void bn_act_kernel(const float* __restrict__ yc,
                              const float* __restrict__ scale,
                              const float* __restrict__ shift,
                              _Float16* __restrict__ out16) {
  int i = blockIdx.x * 256 + threadIdx.x;
  if (i >= N_TOK * D_MODEL) return;
  int d = i & 511;
  float v = yc[i] * scale[d] + shift[d];
  out16[i] = (_Float16)(v * sigmf(v));
}

// ---------------------------------------------------------------------------
// Host orchestration
// ---------------------------------------------------------------------------
static inline void gemm(hipStream_t st, const _Float16* A, long long sA,
                        const _Float16* BT, long long sB, const float* bias,
                        const float* res, long long sR,
                        float* C32, long long sC, _Float16* C16, long long sC16,
                        int M, int N, int K, int lda, int ldb, int ldc,
                        int act, float alpha, int batch) {
  int strips = (M >> 4) * (N >> 6);
  dim3 grid((strips + 3) / 4, batch);
  wmma_gemm_kernel<<<grid, 128, 0, st>>>(A, sA, BT, sB, bias, res, sR, C32, sC,
                                         C16, sC16, M, N, K, lda, ldb, ldc,
                                         act, alpha);
}

extern "C" void kernel_launch(void* const* d_in, const int* in_sizes, int n_in,
                              void* d_out, int out_size, void* d_ws, size_t ws_size,
                              hipStream_t stream) {
  const float* x       = (const float*)d_in[0];
  const float* pos_emb = (const float*)d_in[1];
  const float* ln1_g = (const float*)d_in[2],  * ln1_b = (const float*)d_in[3];
  const float* ln2_g = (const float*)d_in[4],  * ln2_b = (const float*)d_in[5];
  const float* ln3_g = (const float*)d_in[6],  * ln3_b = (const float*)d_in[7];
  const float* ln4_g = (const float*)d_in[8],  * ln4_b = (const float*)d_in[9];
  const float* ln5_g = (const float*)d_in[10], * ln5_b = (const float*)d_in[11];
  const float* ff1_w1 = (const float*)d_in[12], * ff1_b1 = (const float*)d_in[13];
  const float* ff1_w2 = (const float*)d_in[14], * ff1_b2 = (const float*)d_in[15];
  const float* wq = (const float*)d_in[16], * bq = (const float*)d_in[17];
  const float* wk = (const float*)d_in[18], * bk = (const float*)d_in[19];
  const float* wv = (const float*)d_in[20], * bv = (const float*)d_in[21];
  const float* wp = (const float*)d_in[22];
  const float* wo = (const float*)d_in[23], * bo = (const float*)d_in[24];
  const float* pos_u = (const float*)d_in[25], * pos_v = (const float*)d_in[26];
  const float* pw1_w = (const float*)d_in[27], * pw1_b = (const float*)d_in[28];
  const float* dw_w  = (const float*)d_in[29], * dw_b  = (const float*)d_in[30];
  const float* bn_g  = (const float*)d_in[31], * bn_b  = (const float*)d_in[32];
  const float* pw2_w = (const float*)d_in[33], * pw2_b = (const float*)d_in[34];
  const float* ff2_w1 = (const float*)d_in[35], * ff2_b1 = (const float*)d_in[36];
  const float* ff2_w2 = (const float*)d_in[37], * ff2_b2 = (const float*)d_in[38];
  float* out = (float*)d_out;

  // ---- workspace carve-out -------------------------------------------------
  char* w = (char*)d_ws;
  auto carve = [&](size_t bytes) -> void* {
    void* p = (void*)w;
    w += (bytes + 255) & ~(size_t)255;
    return p;
  };
  // All weights stored transposed (N x K) f16 = canonical B^T GEMM layout.
  _Float16* ff1_w1t = (_Float16*)carve((size_t)D_MODEL * D_FF * 2);
  _Float16* ff1_w2t = (_Float16*)carve((size_t)D_FF * D_MODEL * 2);
  _Float16* ff2_w1t = (_Float16*)carve((size_t)D_MODEL * D_FF * 2);
  _Float16* ff2_w2t = (_Float16*)carve((size_t)D_FF * D_MODEL * 2);
  _Float16* wqt = (_Float16*)carve((size_t)D_MODEL * D_MODEL * 2);
  _Float16* wkt = (_Float16*)carve((size_t)D_MODEL * D_MODEL * 2);
  _Float16* wvt = (_Float16*)carve((size_t)D_MODEL * D_MODEL * 2);
  _Float16* wpt = (_Float16*)carve((size_t)D_MODEL * D_MODEL * 2);
  _Float16* wot = (_Float16*)carve((size_t)D_MODEL * D_MODEL * 2);
  _Float16* pw1h = (_Float16*)carve((size_t)2 * D_MODEL * D_MODEL * 2); // already NxK
  _Float16* pw2h = (_Float16*)carve((size_t)D_MODEL * D_MODEL * 2);     // already NxK
  _Float16* posh = (_Float16*)carve((size_t)P_PAD * D_MODEL * 2);
  float*    h      = (float*)carve((size_t)N_TOK * D_MODEL * 4);
  _Float16* t0h    = (_Float16*)carve((size_t)N_TOK * D_MODEL * 2);
  _Float16* hid16  = (_Float16*)carve((size_t)N_TOK * D_FF * 2);
  float*    q32    = (float*)carve((size_t)N_TOK * D_MODEL * 4);
  float*    k32    = (float*)carve((size_t)N_TOK * D_MODEL * 4);
  float*    v32    = (float*)carve((size_t)N_TOK * D_MODEL * 4);
  _Float16* qu16   = (_Float16*)carve((size_t)N_TOK * D_MODEL * 2);
  _Float16* qv16   = (_Float16*)carve((size_t)N_TOK * D_MODEL * 2);
  _Float16* k16    = (_Float16*)carve((size_t)N_TOK * D_MODEL * 2);
  _Float16* vT16   = (_Float16*)carve((size_t)N_TOK * D_MODEL * 2);
  _Float16* p16    = (_Float16*)carve((size_t)P_PAD * D_MODEL * 2);
  float*    scores = (float*)carve((size_t)BATCH_B * N_HEAD * SEQ_T * SEQ_T * 4);
  _Float16* attn16 = (_Float16*)carve((size_t)BATCH_B * N_HEAD * SEQ_T * SEQ_T * 2);
  float*    ao32   = (float*)carve((size_t)N_TOK * D_MODEL * 4);
  _Float16* ao16   = (_Float16*)carve((size_t)N_TOK * D_MODEL * 2);
  float*    zglu   = (float*)carve((size_t)N_TOK * 2 * D_MODEL * 4);
  float*    yg     = (float*)carve((size_t)N_TOK * D_MODEL * 4);
  float*    yc     = (float*)carve((size_t)N_TOK * D_MODEL * 4);
  _Float16* conv16 = (_Float16*)carve((size_t)N_TOK * D_MODEL * 2);
  float*    bn_sc  = (float*)carve(D_MODEL * 4);
  float*    bn_sh  = (float*)carve(D_MODEL * 4);

  auto cvt = [&](const float* s, _Float16* d, int n, int npad) {
    cvt_f16_kernel<<<(npad + 255) / 256, 256, 0, stream>>>(s, d, n, npad);
  };
  auto cvtT = [&](const float* s, _Float16* d, int K, int N) {
    cvt_f16_t_kernel<<<(K * N + 255) / 256, 256, 0, stream>>>(s, d, K, N);
  };
  // ---- weight conversion (transpose to N x K) ------------------------------
  cvtT(ff1_w1, ff1_w1t, D_MODEL, D_FF);
  cvtT(ff1_w2, ff1_w2t, D_FF, D_MODEL);
  cvtT(ff2_w1, ff2_w1t, D_MODEL, D_FF);
  cvtT(ff2_w2, ff2_w2t, D_FF, D_MODEL);
  cvtT(wq, wqt, D_MODEL, D_MODEL);
  cvtT(wk, wkt, D_MODEL, D_MODEL);
  cvtT(wv, wvt, D_MODEL, D_MODEL);
  cvtT(wp, wpt, D_MODEL, D_MODEL);
  cvtT(wo, wot, D_MODEL, D_MODEL);
  cvt(pw1_w, pw1h, 2 * D_MODEL * D_MODEL, 2 * D_MODEL * D_MODEL); // (2D,D)=NxK
  cvt(pw2_w, pw2h, D_MODEL * D_MODEL, D_MODEL * D_MODEL);         // (D,D)=NxK
  cvt(pos_emb, posh, P_POS * D_MODEL, P_PAD * D_MODEL);           // zero-pad row

  const int EW = (N_TOK * D_MODEL + 255) / 256;

  // ---- FFN1: h = x + 0.5 * (silu(ln1(x)@w1+b1) @ w2 + b2) ------------------
  ln_kernel<<<N_TOK, 256, 0, stream>>>(x, ln1_g, ln1_b, t0h, nullptr);
  gemm(stream, t0h, 0, ff1_w1t, 0, ff1_b1, nullptr, 0, nullptr, 0, hid16, 0,
       N_TOK, D_FF, D_MODEL, D_MODEL, D_MODEL, D_FF, /*silu*/1, 1.f, 1);
  gemm(stream, hid16, 0, ff1_w2t, 0, ff1_b2, x, 0, h, 0, nullptr, 0,
       N_TOK, D_MODEL, D_FF, D_FF, D_FF, D_MODEL, 0, 0.5f, 1);

  // ---- Attention -----------------------------------------------------------
  ln_kernel<<<N_TOK, 256, 0, stream>>>(h, ln2_g, ln2_b, t0h, nullptr);
  gemm(stream, t0h, 0, wqt, 0, bq, nullptr, 0, q32, 0, nullptr, 0,
       N_TOK, D_MODEL, D_MODEL, D_MODEL, D_MODEL, D_MODEL, 0, 1.f, 1);
  gemm(stream, t0h, 0, wkt, 0, bk, nullptr, 0, k32, 0, nullptr, 0,
       N_TOK, D_MODEL, D_MODEL, D_MODEL, D_MODEL, D_MODEL, 0, 1.f, 1);
  gemm(stream, t0h, 0, wvt, 0, bv, nullptr, 0, v32, 0, nullptr, 0,
       N_TOK, D_MODEL, D_MODEL, D_MODEL, D_MODEL, D_MODEL, 0, 1.f, 1);
  qkv_prep_kernel<<<EW, 256, 0, stream>>>(q32, k32, v32, pos_u, pos_v,
                                          qu16, qv16, k16, vT16);
  // p = pos_emb @ wp  (padded row is zero); result rows are B^T for bd GEMM
  gemm(stream, posh, 0, wpt, 0, nullptr, nullptr, 0, nullptr, 0, p16, 0,
       P_PAD, D_MODEL, D_MODEL, D_MODEL, D_MODEL, D_MODEL, 0, 1.f, 1);
  // ac: scores = (q+u) @ K^T, batched over 32 (b,h); k16 rows are keys (B^T)
  gemm(stream, qu16, (long long)SEQ_T * D_HEAD, k16, (long long)SEQ_T * D_HEAD,
       nullptr, nullptr, 0, scores, (long long)SEQ_T * SEQ_T, nullptr, 0,
       SEQ_T, SEQ_T, D_HEAD, D_HEAD, D_HEAD, SEQ_T, 0, 1.f, BATCH_B * N_HEAD);
  // bd with fused rel-shift, scatter-add into scores
  {
    int strips = (SEQ_T >> 4) * (P_PAD >> 6);
    dim3 grid((strips + 3) / 4, BATCH_B * N_HEAD);
    bd_shift_kernel<<<grid, 128, 0, stream>>>(qv16, p16, scores);
  }
  softmax_kernel<<<BATCH_B * N_HEAD * SEQ_T, 256, 0, stream>>>(
      scores, attn16, rsqrtf((float)D_HEAD));
  // context = attn @ V; vT16 is (DK x T) per (b,h) == B^T with ldb = T
  gemm(stream, attn16, (long long)SEQ_T * SEQ_T, vT16,
       (long long)D_HEAD * SEQ_T, nullptr, nullptr, 0, ao32,
       (long long)SEQ_T * D_HEAD, nullptr, 0,
       SEQ_T, D_HEAD, SEQ_T, SEQ_T, SEQ_T, D_HEAD, 0, 1.f, BATCH_B * N_HEAD);
  ao_pack_kernel<<<EW, 256, 0, stream>>>(ao32, ao16);
  gemm(stream, ao16, 0, wot, 0, bo, h, 0, h, 0, nullptr, 0,
       N_TOK, D_MODEL, D_MODEL, D_MODEL, D_MODEL, D_MODEL, 0, 1.f, 1);

  // ---- Conv module ---------------------------------------------------------
  ln_kernel<<<N_TOK, 256, 0, stream>>>(h, ln3_g, ln3_b, t0h, nullptr);
  gemm(stream, t0h, 0, pw1h, 0, pw1_b, nullptr, 0, zglu, 0, nullptr, 0,
       N_TOK, 2 * D_MODEL, D_MODEL, D_MODEL, D_MODEL, 2 * D_MODEL, 0, 1.f, 1);
  glu_kernel<<<EW, 256, 0, stream>>>(zglu, yg);
  dwconv_kernel<<<EW, 256, 0, stream>>>(yg, dw_w, dw_b, yc);
  bn_stats_kernel<<<D_MODEL, 256, 0, stream>>>(yc, bn_g, bn_b, bn_sc, bn_sh);
  bn_act_kernel<<<EW, 256, 0, stream>>>(yc, bn_sc, bn_sh, conv16);
  gemm(stream, conv16, 0, pw2h, 0, pw2_b, h, 0, h, 0, nullptr, 0,
       N_TOK, D_MODEL, D_MODEL, D_MODEL, D_MODEL, D_MODEL, 0, 1.f, 1);

  // ---- FFN2 ----------------------------------------------------------------
  ln_kernel<<<N_TOK, 256, 0, stream>>>(h, ln4_g, ln4_b, t0h, nullptr);
  gemm(stream, t0h, 0, ff2_w1t, 0, ff2_b1, nullptr, 0, nullptr, 0, hid16, 0,
       N_TOK, D_FF, D_MODEL, D_MODEL, D_MODEL, D_FF, /*silu*/1, 1.f, 1);
  gemm(stream, hid16, 0, ff2_w2t, 0, ff2_b2, h, 0, h, 0, nullptr, 0,
       N_TOK, D_MODEL, D_FF, D_FF, D_FF, D_MODEL, 0, 0.5f, 1);

  // ---- Final LN ------------------------------------------------------------
  ln_kernel<<<N_TOK, 256, 0, stream>>>(h, ln5_g, ln5_b, nullptr, out);
  (void)in_sizes; (void)n_in; (void)out_size; (void)ws_size;
}
